// FLASH_71038759075984
// MI455X (gfx1250) — matrix operand order repaired and checked by reading, not compiled, not run on hardware
//
#include <hip/hip_runtime.h>

// ---------------------------------------------------------------------------
// Types / WMMA helpers (CDNA5 gfx1250, wave32)
// ---------------------------------------------------------------------------
typedef __bf16 bf16t;
typedef __attribute__((ext_vector_type(16))) __bf16 v16bf;
typedef __attribute__((ext_vector_type(8)))  float  v8f;

__device__ __forceinline__ v8f wmma_bf16(v16bf a, v16bf b, v8f c) {
  // D = A(16x32) * B(32x16) + C(16x16), fp32 accumulate
  return __builtin_amdgcn_wmma_f32_16x16x32_bf16(
      /*neg_a=*/false, a, /*neg_b=*/false, b,
      /*c_mod=*/(short)0, c, /*reuse_a=*/false, /*reuse_b=*/false);
}

__device__ __forceinline__ v8f zero8() {
  v8f z = {0.f, 0.f, 0.f, 0.f, 0.f, 0.f, 0.f, 0.f};
  return z;
}

// A-fragment (16x32 bf16): lane holds row (lane&15), k elements
// { kb..kb+15 }, kb = (lane>=16 ? 16 : 0). Two 16B loads.
__device__ __forceinline__ v16bf load_a32(const bf16t* rowp, int lane) {
  int kb = (lane >> 4) << 4;
  v16bf r;
  *(uint4*)&r         = *(const uint4*)(rowp + kb);
  *(((uint4*)&r) + 1) = *(const uint4*)(rowp + kb + 8);
  return r;
}

// B-fragment (32x16 bf16) where memory holds B TRANSPOSED row-major:
// lane reads 16 contiguous bf16 at row (n = lane&15), k offset (lane>=16)*16.
__device__ __forceinline__ v16bf load_b_row16(const bf16t* p) {
  v16bf r;
  *(uint4*)&r         = *(const uint4*)p;
  *(((uint4*)&r) + 1) = *(const uint4*)(p + 8);
  return r;
}

// ---------------------------------------------------------------------------
// Problem constants
// ---------------------------------------------------------------------------
#define MROWS 16384   // 4 * 4096 tokens
#define DIMC  512
#define HIDC  1024
#define HID2C 2048
#define QKC   128
#define GC    256
#define BGC   64      // batch*groups = 4*16

// ---------------------------------------------------------------------------
// weight fp32 [K][N] -> bf16 transposed [N][K] (one-time)
// ---------------------------------------------------------------------------
__global__ __launch_bounds__(256) void wt_kernel(const float* __restrict__ W,
                                                 bf16t* __restrict__ Wt,
                                                 int K, int N) {
  int idx = blockIdx.x * 256 + threadIdx.x;
  if (idx < K * N) {
    int k = idx / N, n = idx - k * N;
    Wt[(size_t)n * K + k] = (bf16t)W[idx];
  }
}

// ---------------------------------------------------------------------------
// LayerNorm (biased variance) -> bf16, one block per row of 512
// ---------------------------------------------------------------------------
__global__ __launch_bounds__(256) void ln_kernel(const float* __restrict__ x,
                                                 const float* __restrict__ w,
                                                 const float* __restrict__ b,
                                                 bf16t* __restrict__ out) {
  __shared__ float red[256];
  __shared__ float red2[256];
  int tid = threadIdx.x;
  size_t row = blockIdx.x;
  const float* xr = x + row * DIMC;
  float v0 = xr[tid], v1 = xr[tid + 256];
  red[tid]  = v0 + v1;
  red2[tid] = v0 * v0 + v1 * v1;
  __syncthreads();
#pragma unroll
  for (int s = 128; s > 0; s >>= 1) {
    if (tid < s) { red[tid] += red[tid + s]; red2[tid] += red2[tid + s]; }
    __syncthreads();
  }
  float mu  = red[0] * (1.f / DIMC);
  float var = red2[0] * (1.f / DIMC) - mu * mu;
  float rs  = rsqrtf(var + 1e-5f);
  out[row * DIMC + tid]       = (bf16t)((v0 - mu) * rs * w[tid] + b[tid]);
  out[row * DIMC + tid + 256] = (bf16t)((v1 - mu) * rs * w[tid + 256] + b[tid + 256]);
}

// ---------------------------------------------------------------------------
// WMMA GEMM, transposed-operand formulation:
//   C[M tokens][N feat] = X[M][K] @ W[K][N],  Wt = W^T as [N][K].
// A-op = Wt rows (feature = WMMA-M), B-op = X rows (token = WMMA-N).
// Block tile: 128 feat x 128 tok x 64 k; 8 waves -> one 16-feature strip.
// 16 WMMAs per wave per barrier round; grouped loads amortize s_wait_dscnt.
// mode 0: out = bf16(silu(acc+bias));  mode 1: out = fp32(acc+bias+resid)
// ---------------------------------------------------------------------------
__global__ __launch_bounds__(256) void gemm_wmma_kernel(
    const bf16t* __restrict__ Wt, const bf16t* __restrict__ X,
    const float* __restrict__ bias, void* __restrict__ Cout,
    const float* __restrict__ resid, int M, int N, int K, int mode) {
  __shared__ bf16t sW[128 * 64];
  __shared__ bf16t sX[128 * 64];
  int tid = threadIdx.x, lane = tid & 31, wave = tid >> 5;
  int n0 = blockIdx.x * 128;  // feature tile
  int m0 = blockIdx.y * 128;  // token tile

  v8f acc[8];
#pragma unroll
  for (int i = 0; i < 8; ++i) acc[i] = zero8();

  for (int kt = 0; kt < K; kt += 64) {
    __syncthreads();
#pragma unroll
    for (int c = 0; c < 4; ++c) {   // each tile: 128x64 = 1024 chunks of 8
      int idx = tid + c * 256;
      int r = idx >> 3, kc = (idx & 7) << 3;
      *(uint4*)&sW[r * 64 + kc] =
          *(const uint4*)&Wt[(size_t)(n0 + r) * K + kt + kc];
      *(uint4*)&sX[r * 64 + kc] =
          *(const uint4*)&X[(size_t)(m0 + r) * K + kt + kc];
    }
    if (kt + 64 < K) {  // prefetch next k-tile (global_prefetch_b8)
      __builtin_prefetch(
          &Wt[(size_t)(n0 + (tid >> 1)) * K + kt + 64 + ((tid & 1) << 5)], 0, 3);
      __builtin_prefetch(
          &X[(size_t)(m0 + (tid >> 1)) * K + kt + 64 + ((tid & 1) << 5)], 0, 3);
    }
    __syncthreads();
    const bf16t* wrow = &sW[(wave * 16 + (lane & 15)) * 64];
    v16bf a0 = load_a32(wrow, lane);
    v16bf a1 = load_a32(wrow + 32, lane);
#pragma unroll
    for (int nn = 0; nn < 8; nn += 2) {
      const bf16t* x0 = &sX[(nn * 16 + (lane & 15)) * 64 + ((lane >> 4) << 4)];
      const bf16t* x1 = &sX[((nn + 1) * 16 + (lane & 15)) * 64 + ((lane >> 4) << 4)];
      v16bf b00 = load_b_row16(x0);
      v16bf b01 = load_b_row16(x0 + 32);
      v16bf b10 = load_b_row16(x1);
      v16bf b11 = load_b_row16(x1 + 32);
      acc[nn]     = wmma_bf16(a0, b00, acc[nn]);
      acc[nn]     = wmma_bf16(a1, b01, acc[nn]);
      acc[nn + 1] = wmma_bf16(a0, b10, acc[nn + 1]);
      acc[nn + 1] = wmma_bf16(a1, b11, acc[nn + 1]);
    }
  }

  // D(m = feature, n = token): lane -> token, VGPR r -> consecutive features
  int nbase = n0 + wave * 16 + ((lane >> 4) << 3);
  float bsv[8];
#pragma unroll
  for (int r = 0; r < 8; ++r) bsv[r] = bias[nbase + r];
#pragma unroll
  for (int nb = 0; nb < 8; ++nb) {
    size_t m = m0 + nb * 16 + (lane & 15);
    if (mode == 0) {
      bf16t tmp[8];
#pragma unroll
      for (int r = 0; r < 8; ++r) {
        float v = acc[nb][r] + bsv[r];
        tmp[r] = (bf16t)(v / (1.f + __expf(-v)));
      }
      *(uint4*)&((bf16t*)Cout)[m * N + nbase] = *(uint4*)tmp;
    } else {
      float tmp[8];
#pragma unroll
      for (int r = 0; r < 8; ++r)
        tmp[r] = acc[nb][r] + bsv[r] + resid[m * N + nbase + r];
      *(float4*)&((float*)Cout)[m * N + nbase]     = *(float4*)&tmp[0];
      *(float4*)&((float*)Cout)[m * N + nbase + 4] = *(float4*)&tmp[4];
    }
  }
}

// ---------------------------------------------------------------------------
// qk heads: qkh[h] = bf16(qk * gamma[h] + beta[h]); heads 0..3 =
// quad_q, lin_q, quad_k, lin_k
// ---------------------------------------------------------------------------
__global__ __launch_bounds__(256) void qkh_kernel(const bf16t* __restrict__ qk,
                                                  const float* __restrict__ gamma,
                                                  const float* __restrict__ beta,
                                                  bf16t* __restrict__ qkh) {
  size_t idx = (size_t)blockIdx.x * 256 + threadIdx.x;  // < MROWS*QKC
  int d = (int)(idx & (QKC - 1));
  float q = (float)qk[idx];
#pragma unroll
  for (int hh = 0; hh < 4; ++hh)
    qkh[(size_t)hh * MROWS * QKC + idx] =
        (bf16t)(q * gamma[hh * QKC + d] + beta[hh * QKC + d]);
}

// ---------------------------------------------------------------------------
// Attention kernel per (bg = batch*group, et = HID slice of 128).
// LDS: sK [256 tok][128 d] row-major, sVt [128 e][256 tok], sLt [128 d][256 tok],
// sS per-wave [j 32][i 16] scratch -> 204800 B of the 320KB WGP LDS.
// phase 2: lin_kv[d][e] = (lin_k^T V)/G via D = Vt x Lk  (all 16B LDS ops)
// phase 3: S = Q K^T (row reads), O = Vt x S^T (S staged 16B, 16 scalar reads)
// ---------------------------------------------------------------------------
__global__ __launch_bounds__(256) void attn_kernel(
    const bf16t* __restrict__ qkh, const bf16t* __restrict__ h,
    float* __restrict__ linkv, float* __restrict__ acc) {
  extern __shared__ char smem[];
  bf16t* sK  = (bf16t*)smem;         // [256][128]
  bf16t* sVt = sK + GC * QKC;        // [128][256]
  bf16t* sLt = sVt + GC * QKC;       // [128][256]
  bf16t* sS  = sLt + GC * QKC;       // 8 waves * 32*16

  int tid = threadIdx.x, lane = tid & 31, wave = tid >> 5;
  int bg = blockIdx.x, et = blockIdx.y;
  size_t row0 = (size_t)bg * GC;

  const bf16t* qq = qkh;                            // head 0: quad_q
  const bf16t* kq = qkh + (size_t)2 * MROWS * QKC;  // head 2: quad_k
  const bf16t* lk = qkh + (size_t)3 * MROWS * QKC;  // head 3: lin_k

  // staging: K row-major (16B), V/L transposed (scalar stores, one-time)
  for (int c = tid; c < 4096; c += 256) {
    int r = c >> 4, kc = (c & 15) << 3;
    *(uint4*)&sK[r * 128 + kc] = *(const uint4*)&kq[(row0 + r) * QKC + kc];
    uint4 tl = *(const uint4*)&lk[(row0 + r) * QKC + kc];
    uint4 tv = *(const uint4*)&h[(row0 + r) * HID2C + et * 128 + kc];
    const bf16t* pl = (const bf16t*)&tl;
    const bf16t* pv = (const bf16t*)&tv;
#pragma unroll
    for (int e = 0; e < 8; ++e) {
      sLt[(kc + e) * GC + r] = pl[e];
      sVt[(kc + e) * GC + r] = pv[e];
    }
  }
  __syncthreads();

  // ---- phase 2: D(m=e, n=d) = V^T(e,tok) x L(tok,d); wave = one e-strip
  {
    v8f kv[8];
#pragma unroll
    for (int i = 0; i < 8; ++i) kv[i] = zero8();
    for (int kt = 0; kt < GC; kt += 32) {
      v16bf a = load_a32(&sVt[(wave * 16 + (lane & 15)) * GC + kt], lane);
#pragma unroll
      for (int db = 0; db < 8; db += 2) {
        v16bf b0 = load_b_row16(
            &sLt[(db * 16 + (lane & 15)) * GC + kt + ((lane >> 4) << 4)]);
        v16bf b1 = load_b_row16(
            &sLt[((db + 1) * 16 + (lane & 15)) * GC + kt + ((lane >> 4) << 4)]);
        kv[db]     = wmma_bf16(a, b0, kv[db]);
        kv[db + 1] = wmma_bf16(a, b1, kv[db + 1]);
      }
    }
    int ebase = et * 128 + wave * 16 + ((lane >> 4) << 3);
#pragma unroll
    for (int db = 0; db < 8; ++db) {
      int d = db * 16 + (lane & 15);
      float tmp[8];
#pragma unroll
      for (int r = 0; r < 8; ++r) tmp[r] = kv[db][r] * (1.f / GC);
      float* p = &linkv[((size_t)bg * QKC + d) * HIDC + ebase];
      *(float4*)&p[0] = *(float4*)&tmp[0];
      *(float4*)&p[4] = *(float4*)&tmp[4];
    }
  }

  // ---- phase 3: quadratic causal attention, wave handles 16-row strips
  for (int s = wave; s < 16; s += 8) {
    int i0 = s * 16;
    v16bf aqf[4];
#pragma unroll
    for (int c = 0; c < 4; ++c)
      aqf[c] = load_a32(&qq[(row0 + i0 + (lane & 15)) * QKC + c * 32], lane);

    v8f o[8];
#pragma unroll
    for (int i = 0; i < 8; ++i) o[i] = zero8();

    bf16t* myS = sS + wave * 512;  // [j 32][i 16]
    for (int j0 = 0; j0 < i0 + 16; j0 += 32) {
#pragma unroll
      for (int jt = 0; jt < 2; ++jt) {
        const bf16t* kp =
            &sK[(size_t)(j0 + jt * 16 + (lane & 15)) * 128 + ((lane >> 4) << 4)];
        v16bf b0 = load_b_row16(kp);
        v16bf b1 = load_b_row16(kp + 32);
        v16bf b2 = load_b_row16(kp + 64);
        v16bf b3 = load_b_row16(kp + 96);
        v8f sacc = zero8();
        sacc = wmma_bf16(aqf[0], b0, sacc);
        sacc = wmma_bf16(aqf[1], b1, sacc);
        sacc = wmma_bf16(aqf[2], b2, sacc);
        sacc = wmma_bf16(aqf[3], b3, sacc);
        // D(m=i, n=j): lane -> j, r -> consecutive i. Write S^T as [j][i].
        int mibase = (lane >> 4) << 3;
        int jj = j0 + jt * 16 + (lane & 15);
        bf16t tmp[8];
#pragma unroll
        for (int r = 0; r < 8; ++r) {
          float sv = sacc[r] * (1.f / GC);
          sv = fmaxf(sv, 0.f);
          sv *= sv;
          if (jj > i0 + mibase + r) sv = 0.f;  // causal within group
          tmp[r] = (bf16t)sv;
        }
        *(uint4*)&myS[(jt * 16 + (lane & 15)) * 16 + mibase] = *(uint4*)tmp;
      }
      asm volatile("s_wait_dscnt 0" ::: "memory");  // cross-lane LDS RAW
      // B(k=j, n=i) = S^T[j][i] = myS rows: 16 scalar reads per lane
      v16bf bs;
      {
        int ii = lane & 15, jb = (lane >> 4) << 4;
#pragma unroll
        for (int e = 0; e < 16; ++e) bs[e] = myS[(jb + e) * 16 + ii];
      }
#pragma unroll
      for (int eb = 0; eb < 8; eb += 2) {
        v16bf a0 = load_a32(&sVt[(eb * 16 + (lane & 15)) * GC + j0], lane);
        v16bf a1 = load_a32(&sVt[((eb + 1) * 16 + (lane & 15)) * GC + j0], lane);
        o[eb]     = wmma_bf16(a0, bs, o[eb]);
        o[eb + 1] = wmma_bf16(a1, bs, o[eb + 1]);
      }
    }
    // D(m=e, n=i): lane -> token, r -> consecutive e : float4 stores
    size_t tok = row0 + i0 + (lane & 15);
    int ebase = et * 128 + ((lane >> 4) << 3);
#pragma unroll
    for (int eb = 0; eb < 8; ++eb) {
      float tmp[8];
#pragma unroll
      for (int r = 0; r < 8; ++r) tmp[r] = o[eb][r];
      float* p = &acc[tok * HIDC + ebase + eb * 16];
      *(float4*)&p[0] = *(float4*)&tmp[0];
      *(float4*)&p[4] = *(float4*)&tmp[4];
    }
  }
}

// ---------------------------------------------------------------------------
// exclusive cumsum of lin_kv over groups -> bf16 kvexT stored [e][d]
// ---------------------------------------------------------------------------
__global__ __launch_bounds__(256) void kvcum_kernel(const float* __restrict__ linkv,
                                                    bf16t* __restrict__ kvexT) {
  int idx = blockIdx.x * 256 + threadIdx.x;  // < 4*128*1024
  int b  = idx >> 17;
  int de = idx & 131071;     // d*1024 + e
  int d = de >> 10, e = de & 1023;
  float run = 0.f;
#pragma unroll 1
  for (int g = 0; g < 16; ++g) {
    size_t base = ((size_t)(b * 16 + g)) * 131072;
    kvexT[base + (size_t)e * QKC + d] = (bf16t)run;
    run += linkv[base + de];
  }
}

// ---------------------------------------------------------------------------
// lin_out added into acc: D(m=e, n=i) = kv^T(e,d) x lq^T(d,i)
// kv^T rows staged in LDS, lq rows read straight from global (32B loads).
// ---------------------------------------------------------------------------
__global__ __launch_bounds__(256) void linout_kernel(const bf16t* __restrict__ qkh,
                                                     const bf16t* __restrict__ kvexT,
                                                     float* __restrict__ acc) {
  __shared__ bf16t sKV[128 * 128];  // [e-slice][d]
  int tid = threadIdx.x, lane = tid & 31, wave = tid >> 5;
  int bg = blockIdx.x, et = blockIdx.y;
  size_t row0 = (size_t)bg * GC;
  const bf16t* lq = qkh + (size_t)1 * MROWS * QKC;  // head 1: lin_q

  for (int c = tid; c < 2048; c += 256) {
    int er = c >> 4, dc = (c & 15) << 3;
    *(uint4*)&sKV[er * 128 + dc] =
        *(const uint4*)&kvexT[(size_t)bg * 131072 + (size_t)(et * 128 + er) * QKC + dc];
  }
  __syncthreads();

  v16bf al[4];  // A-frags (kv^T), all 4 k-steps, reused across 16 token frags
#pragma unroll
  for (int kt = 0; kt < 4; ++kt)
    al[kt] = load_a32(&sKV[(wave * 16 + (lane & 15)) * 128 + kt * 32], lane);

#pragma unroll 1
  for (int half = 0; half < 2; ++half) {
    v8f o[8];
#pragma unroll
    for (int i = 0; i < 8; ++i) o[i] = zero8();
#pragma unroll
    for (int kt = 0; kt < 4; ++kt) {
#pragma unroll
      for (int nb = 0; nb < 8; nb += 2) {
        v16bf b0 = load_b_row16(
            &lq[(row0 + half * 128 + nb * 16 + (lane & 15)) * QKC + kt * 32 +
                ((lane >> 4) << 4)]);
        v16bf b1 = load_b_row16(
            &lq[(row0 + half * 128 + (nb + 1) * 16 + (lane & 15)) * QKC + kt * 32 +
                ((lane >> 4) << 4)]);
        o[nb]     = wmma_bf16(al[kt], b0, o[nb]);
        o[nb + 1] = wmma_bf16(al[kt], b1, o[nb + 1]);
      }
    }
    int ebase = et * 128 + wave * 16 + ((lane >> 4) << 3);
#pragma unroll
    for (int nb = 0; nb < 8; ++nb) {
      size_t tok = row0 + half * 128 + nb * 16 + (lane & 15);
      float* p = &acc[tok * HIDC + ebase];
      float4 t0 = *(float4*)&p[0];
      float4 t1 = *(float4*)&p[4];
      t0.x += o[nb][0]; t0.y += o[nb][1]; t0.z += o[nb][2]; t0.w += o[nb][3];
      t1.x += o[nb][4]; t1.y += o[nb][5]; t1.z += o[nb][6]; t1.w += o[nb][7];
      *(float4*)&p[0] = t0;
      *(float4*)&p[4] = t1;
    }
  }
}

// ---------------------------------------------------------------------------
// gate multiply: afinal = bf16(gate * (quad_out + lin_out))
// ---------------------------------------------------------------------------
__global__ __launch_bounds__(256) void gatemul_kernel(const bf16t* __restrict__ h,
                                                      const float* __restrict__ acc,
                                                      bf16t* __restrict__ afinal) {
  size_t idx = (size_t)blockIdx.x * 256 + threadIdx.x;  // < MROWS*HIDC
  size_t row = idx >> 10;
  int e = (int)(idx & 1023);
  float gate = (float)h[row * HID2C + HIDC + e];
  afinal[idx] = (bf16t)(gate * acc[idx]);
}

// ---------------------------------------------------------------------------
// Host launcher
// ---------------------------------------------------------------------------
extern "C" void kernel_launch(void* const* d_in, const int* in_sizes, int n_in,
                              void* d_out, int out_size, void* d_ws, size_t ws_size,
                              hipStream_t stream) {
  const float* x     = (const float*)d_in[0];
  const float* ln_w  = (const float*)d_in[1];
  const float* ln_b  = (const float*)d_in[2];
  const float* Wh    = (const float*)d_in[3];
  const float* bh    = (const float*)d_in[4];
  const float* Wqk   = (const float*)d_in[5];
  const float* bqk   = (const float*)d_in[6];
  const float* gamma = (const float*)d_in[7];
  const float* beta  = (const float*)d_in[8];
  const float* Wo    = (const float*)d_in[9];
  const float* bo    = (const float*)d_in[10];

  char* ws = (char*)d_ws;
  size_t off = 0;
  auto carve = [&](size_t bytes) {
    char* p = ws + off;
    off += (bytes + 255) & ~(size_t)255;
    return p;
  };
  bf16t* normed = (bf16t*)carve((size_t)MROWS * DIMC * 2);
  bf16t* wh_t   = (bf16t*)carve((size_t)DIMC * HID2C * 2);   // [2048][512]
  bf16t* wqk_t  = (bf16t*)carve((size_t)DIMC * QKC * 2);     // [128][512]
  bf16t* wo_t   = (bf16t*)carve((size_t)HIDC * DIMC * 2);    // [512][1024]
  bf16t* hbuf   = (bf16t*)carve((size_t)MROWS * HID2C * 2);  // [v | gate]
  bf16t* qkbuf  = (bf16t*)carve((size_t)MROWS * QKC * 2);
  bf16t* qkh    = (bf16t*)carve((size_t)4 * MROWS * QKC * 2);
  float* linkv  = (float*)carve((size_t)BGC * QKC * HIDC * 4);
  bf16t* kvexT  = (bf16t*)carve((size_t)BGC * QKC * HIDC * 2);
  float* accb   = (float*)carve((size_t)MROWS * HIDC * 4);
  bf16t* afin   = (bf16t*)carve((size_t)MROWS * HIDC * 2);

  const int attn_lds = (3 * GC * QKC + 8 * 512) * 2;  // 204800 B
  hipFuncSetAttribute((const void*)attn_kernel,
                      hipFuncAttributeMaxDynamicSharedMemorySize, attn_lds);

  // 1. weights -> bf16, transposed
  wt_kernel<<<(DIMC * HID2C + 255) / 256, 256, 0, stream>>>(Wh, wh_t, DIMC, HID2C);
  wt_kernel<<<(DIMC * QKC + 255) / 256, 256, 0, stream>>>(Wqk, wqk_t, DIMC, QKC);
  wt_kernel<<<(HIDC * DIMC + 255) / 256, 256, 0, stream>>>(Wo, wo_t, HIDC, DIMC);

  // 2. LayerNorm -> bf16 normed
  ln_kernel<<<MROWS, 256, 0, stream>>>(x, ln_w, ln_b, normed);

  // 3. h = silu(normed @ Wh + bh)  [M=16384, Nfeat=2048, K=512]
  gemm_wmma_kernel<<<dim3(HID2C / 128, MROWS / 128), 256, 0, stream>>>(
      wh_t, normed, bh, (void*)hbuf, nullptr, MROWS, HID2C, DIMC, 0);

  // 4. qk = silu(normed @ Wqk + bqk)  [Nfeat=128]
  gemm_wmma_kernel<<<dim3(QKC / 128, MROWS / 128), 256, 0, stream>>>(
      wqk_t, normed, bqk, (void*)qkbuf, nullptr, MROWS, QKC, DIMC, 0);

  // 5. 4-head offset/scale
  qkh_kernel<<<(MROWS * QKC) / 256, 256, 0, stream>>>(qkbuf, gamma, beta, qkh);

  // 6. quadratic attention + per-group lin_kv
  attn_kernel<<<dim3(BGC, HIDC / 128), 256, attn_lds, stream>>>(qkh, hbuf, linkv, accb);

  // 7. exclusive cumsum of lin_kv over groups (writes transposed [e][d])
  kvcum_kernel<<<(4 * QKC * HIDC) / 256, 256, 0, stream>>>(linkv, kvexT);

  // 8. lin_out added into acc
  linout_kernel<<<dim3(BGC, HIDC / 128), 256, 0, stream>>>(qkh, kvexT, accb);

  // 9. gate * (quad + lin) -> bf16
  gatemul_kernel<<<(MROWS * HIDC) / 256, 256, 0, stream>>>(hbuf, accb, afin);

  // 10. out = afin @ Wo + bo + x  [Nfeat=512, K=1024], fp32 out
  gemm_wmma_kernel<<<dim3(DIMC / 128, MROWS / 128), 256, 0, stream>>>(
      wo_t, afin, bo, d_out, x, MROWS, DIMC, HIDC, 1);

  (void)in_sizes; (void)n_in; (void)out_size; (void)ws_size;
}
